// LoRAQKVParallelLinear_11295763988854
// MI455X (gfx1250) — matrix operand order repaired and checked by reading, not compiled, not run on hardware
//
#include <hip/hip_runtime.h>

// ---------------------------------------------------------------------------
// LoRA QKV fused projection for MI455X (gfx1250, wave32, WMMA bf16)
//   out[T,6144] = x[T,4096] @ W[6144,4096]^T  +  LoRA(Q/K/V) delta
// Main GEMM: v_wmma_f32_16x16x32_bf16, 256x128x32 block tile, 64x64 wave tile,
// double-buffered LDS fed by async global->LDS copies (ASYNCcnt pipeline),
// immediate-offset addressing so each stage uses one LDS VGPR + one addr pair.
// ---------------------------------------------------------------------------

typedef __bf16  v8s  __attribute__((ext_vector_type(8)));    // 16 bytes
typedef __bf16  v16s __attribute__((ext_vector_type(16)));   // 32 bytes (A/B frag)
typedef float   v8f  __attribute__((ext_vector_type(8)));    // C/D frag
typedef float   v4f  __attribute__((ext_vector_type(4)));

union Frag { v16s v; v8s h[2]; };

#define T_TOK   8192
#define HID     4096
#define NOUT    6144   // 4096 + 1024 + 1024
#define Q_SIZE  4096
#define KV_SIZE 1024
#define RANK    16

#define BM 256
#define BN 128
#define BK 32
#define LDS_STRIDE 40   // bf16 elements per LDS row (32 + 8 pad -> 80B rows)

// ---- CDNA5 async global->LDS copy (ASYNCcnt), 16 bytes per lane ------------
// INST_OFFSET applies to BOTH the LDS dest (VDST + off)
// and the global source (VADDR + off), so one base VGPR pair serves a whole
// row chunk via offset immediates.
template <int OFF>
__device__ __forceinline__ void async_copy_b128(unsigned lds_byte_off,
                                                const void* gptr) {
    asm volatile("global_load_async_to_lds_b128 %0, %1, off offset:%2"
                 :: "v"(lds_byte_off), "v"(gptr), "i"(OFF) : "memory");
}
__device__ __forceinline__ void wait_async0() {
    asm volatile("s_wait_asynccnt 0x0" ::: "memory");
}

// ---------------------------------------------------------------------------
// 1) f32 -> bf16 streaming convert (8 elems / thread, b128 in / b128 out)
// ---------------------------------------------------------------------------
__global__ void lqkv_cvt_bf16(const float* __restrict__ in,
                              __bf16* __restrict__ out, long n) {
    long i = ((long)blockIdx.x * blockDim.x + threadIdx.x) * 8;
    if (i + 8 > n) return;
    v4f x0 = *(const v4f*)(in + i);
    v4f x1 = *(const v4f*)(in + i + 4);
    v8s o;
    o[0] = (__bf16)x0[0]; o[1] = (__bf16)x0[1];
    o[2] = (__bf16)x0[2]; o[3] = (__bf16)x0[3];
    o[4] = (__bf16)x1[0]; o[5] = (__bf16)x1[1];
    o[6] = (__bf16)x1[2]; o[7] = (__bf16)x1[3];
    *(v8s*)(out + i) = o;
}

// ---------------------------------------------------------------------------
// 2) LoRA down-projection: a[t, i*16+r] = scaling[slot] * dot(x[t], A[slot,i,r])
// ---------------------------------------------------------------------------
__global__ __launch_bounds__(256)
void lqkv_lora_a(const float* __restrict__ x,
                 const float* __restrict__ loraA,   // [S,3,R,HID]
                 const float* __restrict__ scaling, // [S]
                 const int*   __restrict__ t2s,     // [T]
                 float* __restrict__ aout)          // [T, 48]
{
    __shared__ float xs[HID];
    const int t    = blockIdx.x;
    const int slot = t2s[t];
    const float sc = scaling[slot];

    const float* xr = x + (size_t)t * HID;
    for (int h = threadIdx.x * 4; h < HID; h += 256 * 4)
        *(v4f*)&xs[h] = *(const v4f*)(xr + h);
    __syncthreads();

    const int wave = threadIdx.x >> 5;
    const int lane = threadIdx.x & 31;

    for (int o = wave; o < 48; o += 8) {           // o = i*16 + r
        const float* Ar = loraA + ((size_t)slot * 48 + o) * HID;
        float acc = 0.f;
        for (int h = lane * 4; h < HID; h += 32 * 4) {
            v4f xv = *(const v4f*)&xs[h];
            v4f av = *(const v4f*)(Ar + h);
            acc += xv[0]*av[0] + xv[1]*av[1] + xv[2]*av[2] + xv[3]*av[3];
        }
        #pragma unroll
        for (int off = 16; off > 0; off >>= 1)
            acc += __shfl_down(acc, off, 32);
        if (lane == 0) aout[(size_t)t * 48 + o] = acc * sc;
    }
}

// ---------------------------------------------------------------------------
// 3) Main GEMM (bf16 WMMA, f32 accum) with fused LoRA-B epilogue.
//    Block: 256 thr = 8 waves, tile 256x128. Wave grid 4(M)x2(N),
//    each wave: 4x4 tiles of 16x16 -> 16 accumulators, 16 WMMA / k-step.
//    Staging: async global->LDS b128 with offset immediates, double buffered.
// ---------------------------------------------------------------------------
__global__ __launch_bounds__(256)
void lqkv_gemm_wmma(const __bf16* __restrict__ Xb,   // [T, HID] bf16
                    const __bf16* __restrict__ Wb,   // [NOUT, HID] bf16
                    const float*  __restrict__ aA,   // [T, 48]
                    const float*  __restrict__ Bq,   // [S, Q_SIZE, R]
                    const float*  __restrict__ Bk,   // [S, KV_SIZE, R]
                    const float*  __restrict__ Bv,   // [S, KV_SIZE, R]
                    const int*    __restrict__ t2s,  // [T]
                    float* __restrict__ out)         // [T, NOUT]
{
    __shared__ __bf16 As[2][BM * LDS_STRIDE];   // 2 x 20KB
    __shared__ __bf16 Bs[2][BN * LDS_STRIDE];   // 2 x 10KB

    const int tileN = blockIdx.x * BN;
    const int tileM = blockIdx.y * BM;
    const int tid   = threadIdx.x;

    // A staging: 1 thread per row (256 rows), 32 elems = 4x async b128
    const __bf16* gA = Xb + (size_t)(tileM + tid) * HID;
    // B staging: 2 threads per row (128 rows), 16 elems each = 2x async b128
    const __bf16* gB = Wb + (size_t)(tileN + (tid >> 1)) * HID + (tid & 1) * 16;
    const unsigned ldsA = (unsigned)(size_t)&As[0][tid * LDS_STRIDE];
    const unsigned ldsB = (unsigned)(size_t)&Bs[0][(tid >> 1) * LDS_STRIDE + (tid & 1) * 16];
    const unsigned bufStrideA = (unsigned)(BM * LDS_STRIDE * sizeof(__bf16));
    const unsigned bufStrideB = (unsigned)(BN * LDS_STRIDE * sizeof(__bf16));

    const int wave = tid >> 5;
    const int lane = tid & 31;
    const int wm   = wave & 3;              // M strip: 64 rows
    const int wn   = wave >> 2;             // N strip: 64 cols
    const int mrow = lane & 15;             // frag row / col within 16
    const int half = lane >> 4;             // 0/1

    v8f acc[4][4] = {};

    // prologue: async-stage buffer 0 (k = 0)
    {
        async_copy_b128<0>(ldsA, gA);
        async_copy_b128<16>(ldsA, gA);
        async_copy_b128<32>(ldsA, gA);
        async_copy_b128<48>(ldsA, gA);
        async_copy_b128<0>(ldsB, gB);
        async_copy_b128<16>(ldsB, gB);
        wait_async0();
    }
    __syncthreads();

    const int KT = HID / BK;   // 128 k-steps
    for (int kt = 0; kt < KT; ++kt) {
        const int buf = kt & 1;
        const bool more = (kt + 1 < KT);

        // --- kick off async staging of tile k+1 into the other buffer ---
        if (more) {
            const long ko = (long)(kt + 1) * BK;
            const __bf16*  ga = gA + ko;                       // one addr pair
            const __bf16*  gb = gB + ko;
            const unsigned la = ldsA + (buf ^ 1) * bufStrideA; // one LDS VGPR
            const unsigned lb = ldsB + (buf ^ 1) * bufStrideB;
            async_copy_b128<0>(la, ga);
            async_copy_b128<16>(la, ga);
            async_copy_b128<32>(la, ga);
            async_copy_b128<48>(la, ga);
            async_copy_b128<0>(lb, gb);
            async_copy_b128<16>(lb, gb);
            if (kt + 2 < KT) {
                __builtin_prefetch(gA + ko + BK, 0, 0);
                __builtin_prefetch(gB + ko + BK, 0, 0);
            }
        }

        // --- fragment loads from LDS (contiguous k = 16h + s, same for A and B) ---
        Frag af[4], bf[4];
        #pragma unroll
        for (int mi = 0; mi < 4; ++mi) {
            const __bf16* pa = &As[buf][(wm * 64 + mi * 16 + mrow) * LDS_STRIDE];
            af[mi].h[0] = *(const v8s*)(pa + 16 * half);       // K = 16h .. 16h+7
            af[mi].h[1] = *(const v8s*)(pa + 16 * half + 8);   // K = 16h+8 ..
        }
        #pragma unroll
        for (int ni = 0; ni < 4; ++ni) {
            const __bf16* pb = &Bs[buf][(wn * 64 + ni * 16 + mrow) * LDS_STRIDE];
            bf[ni].h[0] = *(const v8s*)(pb + 16 * half);       // K = 16h .. 16h+7
            bf[ni].h[1] = *(const v8s*)(pb + 16 * half + 8);   // K = 16h+8 ..
        }

        // --- 16 WMMAs (64x64 wave tile, one 32-K step) ---
        #pragma unroll
        for (int mi = 0; mi < 4; ++mi)
            #pragma unroll
            for (int ni = 0; ni < 4; ++ni)
                acc[mi][ni] = __builtin_amdgcn_wmma_f32_16x16x32_bf16(
                    false, af[mi].v, false, bf[ni].v,
                    (short)0, acc[mi][ni], false, false);

        // --- drain this wave's async copies, then block-wide handoff ---
        if (more) wait_async0();
        __syncthreads();
    }

    // --- fused LoRA-B epilogue: out = acc + a[t,seg,:] . B_seg[slot,col,:] ---
    #pragma unroll
    for (int ni = 0; ni < 4; ++ni) {
        const int gcol = tileN + wn * 64 + ni * 16 + mrow;
        int seg, bcol;
        const float* Bsel;
        if (gcol < Q_SIZE)               { seg = 0; Bsel = Bq; bcol = gcol; }
        else if (gcol < Q_SIZE + KV_SIZE){ seg = 1; Bsel = Bk; bcol = gcol - Q_SIZE; }
        else                             { seg = 2; Bsel = Bv; bcol = gcol - (Q_SIZE + KV_SIZE); }
        const size_t dsz = (seg == 0) ? Q_SIZE : KV_SIZE;

        #pragma unroll
        for (int mi = 0; mi < 4; ++mi) {
            #pragma unroll
            for (int vr = 0; vr < 8; ++vr) {
                const int grow = tileM + wm * 64 + mi * 16 + vr + 8 * half;
                const int slot = t2s[grow];
                const v4f* br = (const v4f*)(Bsel + ((size_t)slot * dsz + bcol) * RANK);
                const v4f* ar = (const v4f*)(aA + (size_t)grow * 48 + seg * RANK);
                float d = 0.f;
                #pragma unroll
                for (int q = 0; q < 4; ++q) {
                    v4f a4 = ar[q], b4 = br[q];
                    d += a4[0]*b4[0] + a4[1]*b4[1] + a4[2]*b4[2] + a4[3]*b4[3];
                }
                out[(size_t)grow * NOUT + gcol] = acc[mi][ni][vr] + d;
            }
        }
    }
}

// ---------------------------------------------------------------------------
// launch
// ---------------------------------------------------------------------------
extern "C" void kernel_launch(void* const* d_in, const int* in_sizes, int n_in,
                              void* d_out, int out_size, void* d_ws, size_t ws_size,
                              hipStream_t stream) {
    const float* x       = (const float*)d_in[0];
    const float* W       = (const float*)d_in[1];
    const float* loraA   = (const float*)d_in[2];
    const float* Bq      = (const float*)d_in[3];
    const float* Bk      = (const float*)d_in[4];
    const float* Bv      = (const float*)d_in[5];
    const float* scaling = (const float*)d_in[6];
    const int*   t2s     = (const int*)d_in[7];
    float*       out     = (float*)d_out;
    (void)in_sizes; (void)n_in; (void)out_size; (void)ws_size;

    // workspace: xbf (64MB) | wbf (48MB) | aA (1.5MB)
    __bf16* xbf = (__bf16*)d_ws;
    __bf16* wbf = xbf + (size_t)T_TOK * HID;
    float*  aA  = (float*)(wbf + (size_t)NOUT * HID);

    const long nx = (long)T_TOK * HID;   // 33,554,432
    const long nw = (long)NOUT * HID;    // 25,165,824
    lqkv_cvt_bf16<<<(int)(nx / (256 * 8)), 256, 0, stream>>>(x, xbf, nx);
    lqkv_cvt_bf16<<<(int)(nw / (256 * 8)), 256, 0, stream>>>(W, wbf, nw);

    lqkv_lora_a<<<T_TOK, 256, 0, stream>>>(x, loraA, scaling, t2s, aA);

    dim3 grid(NOUT / BN, T_TOK / BM);    // 48 x 32
    lqkv_gemm_wmma<<<grid, 256, 0, stream>>>(xbf, wbf, aA, Bq, Bk, Bv, t2s, out);
}
